// MoEElementFusion_28278064677439
// MI455X (gfx1250) — matrix operand order repaired and checked by reading, not compiled, not run on hardware
//
#include <hip/hip_runtime.h>
#include <hip/hip_bf16.h>
#include <math.h>

// Problem constants (from reference)
#define BB 4
#define LL 512
#define DD 1024
#define EE 16
#define TK 4
#define FF 4096
#define NN 2048          // BB*LL tokens
#define MAX_TILES 128    // NN/16 worst-case tiles per expert

typedef __attribute__((ext_vector_type(16))) __bf16          v16bf;
typedef __attribute__((ext_vector_type(8)))  float           v8f;
typedef __attribute__((ext_vector_type(16))) unsigned short  us16;
typedef __attribute__((ext_vector_type(8)))  unsigned short  us8;

union BF16x16 { us16 u; v16bf v; };

// f32 -> bf16, round-to-nearest-even
__device__ __forceinline__ unsigned short f2bf(float f) {
    unsigned int u = __float_as_uint(f);
    u = (u + 0x7FFFu + ((u >> 16) & 1u)) >> 16;
    return (unsigned short)u;
}

__device__ __forceinline__ void pack4(us16& u, int base, float4 q) {
    u[base + 0] = f2bf(q.x);
    u[base + 1] = f2bf(q.y);
    u[base + 2] = f2bf(q.z);
    u[base + 3] = f2bf(q.w);
}

__device__ __forceinline__ float gelu_exact(float v) {
    return 0.5f * v * (1.0f + erff(v * 0.70710678118654752f));
}

// B-fragment loaders: 16 values per lane (K-contiguous, k = 16h + s, same order as the A fragments).
__device__ __forceinline__ v16bf load_bfrag_bf16(const unsigned short* p) {
    us8 lo = *reinterpret_cast<const us8*>(p);
    us8 hi = *reinterpret_cast<const us8*>(p + 8);
    BF16x16 fb;
    fb.u = __builtin_shufflevector(lo, hi, 0,1,2,3,4,5,6,7,8,9,10,11,12,13,14,15);
    return fb.v;
}
__device__ __forceinline__ v16bf load_bfrag_f32(const float* p) {
    const float4* w4 = reinterpret_cast<const float4*>(p);
    BF16x16 fb;
    pack4(fb.u, 0, w4[0]); pack4(fb.u, 4, w4[1]);
    pack4(fb.u, 8, w4[2]); pack4(fb.u, 12, w4[3]);
    return fb.v;
}

// ---------------------------------------------------------------------------
// Kernel 0: bulk f32 -> bf16 weight conversion (pure bandwidth, once/launch).
// Amortizes conversion ~32x vs converting per GEMM use, and halves the
// weight bytes the GEMM inner loop pulls from HBM/L2.
// ---------------------------------------------------------------------------
__global__ __launch_bounds__(256)
void cvt_f32_to_bf16(const float* __restrict__ src, unsigned short* __restrict__ dst,
                     int n8) {                     // n8 = element count / 8
    int i = blockIdx.x * 256 + threadIdx.x;
    if (i >= n8) return;
    const float4* s4 = reinterpret_cast<const float4*>(src);
    float4 a = s4[i * 2 + 0];
    float4 b = s4[i * 2 + 1];
    us8 o;
    o[0] = f2bf(a.x); o[1] = f2bf(a.y); o[2] = f2bf(a.z); o[3] = f2bf(a.w);
    o[4] = f2bf(b.x); o[5] = f2bf(b.y); o[6] = f2bf(b.z); o[7] = f2bf(b.w);
    reinterpret_cast<us8*>(dst)[i] = o;
}

// ---------------------------------------------------------------------------
// Kernel 1: Laplace-gate router + top-4 + softmax + per-expert compaction.
// One wave32 per token; 8 tokens per 256-thread block.
// ---------------------------------------------------------------------------
__global__ __launch_bounds__(256)
void router_topk(const float* __restrict__ x, const float* __restrict__ keys,
                 const float* __restrict__ rw, const float* __restrict__ rb,
                 int* __restrict__ counts, int* __restrict__ pkArr,
                 float* __restrict__ gateArr) {
    const int lane = threadIdx.x & 31;
    const int t    = blockIdx.x * 8 + (threadIdx.x >> 5);

    const float* xr = x + (size_t)t * DD;
    float xv[32];
#pragma unroll
    for (int j = 0; j < 32; ++j) xv[j] = xr[j * 32 + lane];

    float lg[EE];
    for (int e = 0; e < EE; ++e) {
        const float* kr = keys + (size_t)e * DD;
        const float* rr = rw   + (size_t)e * DD;
        float d2 = 0.0f, dt = 0.0f;
#pragma unroll
        for (int j = 0; j < 32; ++j) {
            float kv = kr[j * 32 + lane];
            float rv = rr[j * 32 + lane];
            float df = xv[j] - kv;
            d2 = fmaf(df, df, d2);
            dt = fmaf(xv[j], rv, dt);
        }
#pragma unroll
        for (int off = 16; off > 0; off >>= 1) {
            d2 += __shfl_xor(d2, off, 32);
            dt += __shfl_xor(dt, off, 32);
        }
        lg[e] = -d2 + dt + rb[e];   // identical in all lanes after reduction
    }

    // Lane-uniform top-4 (ties -> lowest index, matching lax.top_k)
    float tv[TK]; int ti[TK]; unsigned mask = 0u;
#pragma unroll
    for (int k = 0; k < TK; ++k) {
        float best = -3.4e38f; int bi = 0;
        for (int e = 0; e < EE; ++e)
            if (!((mask >> e) & 1u) && lg[e] > best) { best = lg[e]; bi = e; }
        mask |= 1u << bi; tv[k] = best; ti[k] = bi;
    }
    float m = tv[0], s = 0.0f, wv[TK];
#pragma unroll
    for (int k = 0; k < TK; ++k) { wv[k] = expf(tv[k] - m); s += wv[k]; }
    float inv = 1.0f / s;

    if (lane == 0) {
#pragma unroll
        for (int k = 0; k < TK; ++k) {
            int e = ti[k];
            int pos = atomicAdd(&counts[e], 1);
            pkArr[e * NN + pos]   = (t << 2) | k;   // token*4 + slot
            gateArr[e * NN + pos] = wv[k] * inv;
        }
    }
}

// ---------------------------------------------------------------------------
// Kernel 2: grouped per-expert fused FC1 -> GELU -> FC2 on 16-token tiles.
// 256 threads = 8 wave32s; bf16 WMMA 16x16x32, f32 accumulate.
// BF16W: weights pre-converted to bf16 in ws (no VALU in the inner loop).
// ---------------------------------------------------------------------------
template<bool BF16W>
__global__ __launch_bounds__(256)
void moe_expert_gemm(const float* __restrict__ x,
                     const float* __restrict__ w1f, const float* __restrict__ b1,
                     const float* __restrict__ w2f, const float* __restrict__ b2,
                     const unsigned short* __restrict__ w1b,
                     const unsigned short* __restrict__ w2b,
                     const int* __restrict__ counts,
                     const int* __restrict__ pkArr, const float* __restrict__ gateArr,
                     float* __restrict__ outb) {
    const int e    = blockIdx.x >> 7;     // / MAX_TILES (expert-major for L2 reuse)
    const int tile = blockIdx.x & 127;
    const int cnt  = counts[e];
    if (tile * 16 >= cnt) return;         // uniform early exit

    // Padded strides: 516 dw % 64 = 4 and 132 dw % 64 = 4 -> conflict-free rows.
    __shared__ unsigned short Xs[16 * 1032];
    __shared__ unsigned short Hs[16 * 264];
    __shared__ int   sPk[16];
    __shared__ float sGate[16];
    __shared__ int   sValid[16];

    const int tid = threadIdx.x;
    if (tid < 16) {
        int idx = tile * 16 + tid;
        if (idx < cnt) {
            sPk[tid] = pkArr[e * NN + idx];
            sGate[tid] = gateArr[e * NN + idx];
            sValid[tid] = 1;
        } else {
            sPk[tid] = 0; sGate[tid] = 0.0f; sValid[tid] = 0;
        }
    }
    __syncthreads();

    // Stage X tile (16 x 1024 f32 -> bf16 LDS), 16 float4 per thread.
#pragma unroll
    for (int it = 0; it < 16; ++it) {
        int pos = it * 256 + tid;
        int row = pos >> 8, c4 = pos & 255;
        int token = sPk[row] >> 2;
        float4 v = reinterpret_cast<const float4*>(x + (size_t)token * DD)[c4];
        unsigned short* dp = &Xs[row * 1032 + c4 * 4];
        dp[0] = f2bf(v.x); dp[1] = f2bf(v.y); dp[2] = f2bf(v.z); dp[3] = f2bf(v.w);
    }
    __syncthreads();

    const int wave = tid >> 5, lane = tid & 31;
    const int lr = lane & 15, h = lane >> 4;

    const float*          w1fe = w1f + (size_t)e * FF * DD;
    const float*          w2fe = w2f + (size_t)e * DD * FF;
    const unsigned short* w1be = w1b + (size_t)e * FF * DD;
    const unsigned short* w2be = w2b + (size_t)e * DD * FF;
    const float*          b1e  = b1  + (size_t)e * FF;

    v8f acc[8] = {};   // wave owns output columns [wave*128, wave*128+128)

#pragma unroll 1
    for (int sc = 0; sc < 16; ++sc) {      // super-chunks of 256 F-values
        const int f0 = sc * 256;
        const int fA = f0 + wave * 32;     // this wave's 32-wide H chunk

        // ---- Phase A: H = gelu(X @ W1_chunk^T + b1), 16x32, per wave ----
        v8f ha0 = {}, ha1 = {};
        const unsigned short* xrow = &Xs[lr * 1032];
#pragma unroll 4
        for (int kk = 0; kk < DD; kk += 32) {
            // A fragment: row lr, K run [kk+h*16,+16)
            BF16x16 fa;
            us8 alo = *reinterpret_cast<const us8*>(xrow + kk + h * 16);
            us8 ahi = *reinterpret_cast<const us8*>(xrow + kk + h * 16 + 8);
            fa.u = __builtin_shufflevector(alo, ahi, 0,1,2,3,4,5,6,7,8,9,10,11,12,13,14,15);

            v16bf fb0, fb1;
            if constexpr (BF16W) {
                const unsigned short* p0 = w1be + (size_t)(fA + lr) * DD + kk + h * 16;
                const unsigned short* p1 = w1be + (size_t)(fA + 16 + lr) * DD + kk + h * 16;
                __builtin_prefetch(p0 + 64, 0, 1);          // next-K prefetch
                fb0 = load_bfrag_bf16(p0);
                fb1 = load_bfrag_bf16(p1);
            } else {
                const float* p0 = w1fe + (size_t)(fA + lr) * DD + kk + h * 16;
                const float* p1 = w1fe + (size_t)(fA + 16 + lr) * DD + kk + h * 16;
                __builtin_prefetch(p0 + 32, 0, 1);
                fb0 = load_bfrag_f32(p0);
                fb1 = load_bfrag_f32(p1);
            }
            ha0 = __builtin_amdgcn_wmma_f32_16x16x32_bf16(
                      false, fa.v, false, fb0, (short)0, ha0, false, false);
            ha1 = __builtin_amdgcn_wmma_f32_16x16x32_bf16(
                      false, fa.v, false, fb1, (short)0, ha1, false, false);
        }
        // epilogue: +b1, exact GELU, bf16 -> shared H
        {
            float bb0 = b1e[fA + lr];
            float bb1 = b1e[fA + 16 + lr];
#pragma unroll
            for (int i = 0; i < 8; ++i) {
                int mrow = h * 8 + i;          // C/D layout: M = h*8 + vgpr index
                Hs[mrow * 264 + wave * 32 + lr]      = f2bf(gelu_exact(ha0[i] + bb0));
                Hs[mrow * 264 + wave * 32 + 16 + lr] = f2bf(gelu_exact(ha1[i] + bb1));
            }
        }
        __syncthreads();   // H chunk visible to all waves

        // ---- Phase B: acc += H(16x256) @ W2_chunk^T for this wave's 128 cols ----
        const unsigned short* hrow = &Hs[lr * 264];
#pragma unroll 2
        for (int ks = 0; ks < 8; ++ks) {
            int kkB = ks * 32;
            BF16x16 fa;
            us8 alo = *reinterpret_cast<const us8*>(hrow + kkB + h * 16);
            us8 ahi = *reinterpret_cast<const us8*>(hrow + kkB + h * 16 + 8);
            fa.u = __builtin_shufflevector(alo, ahi, 0,1,2,3,4,5,6,7,8,9,10,11,12,13,14,15);
            const int fbase = f0 + kkB + h * 16;
#pragma unroll
            for (int ct = 0; ct < 8; ++ct) {
                int d = wave * 128 + ct * 16 + lr;
                v16bf fb;
                if constexpr (BF16W) {
                    fb = load_bfrag_bf16(w2be + (size_t)d * FF + fbase);
                } else {
                    fb = load_bfrag_f32(w2fe + (size_t)d * FF + fbase);
                }
                acc[ct] = __builtin_amdgcn_wmma_f32_16x16x32_bf16(
                              false, fa.v, false, fb, (short)0, acc[ct], false, false);
            }
        }
        __syncthreads();   // protect Hs before next super-chunk overwrites it
    }

    // Store gated results per (token, slot) -> out_buf[(t*4+k)*D + d]
#pragma unroll
    for (int ct = 0; ct < 8; ++ct) {
        int d = wave * 128 + ct * 16 + lr;
        float b2v = b2[e * DD + d];
#pragma unroll
        for (int i = 0; i < 8; ++i) {
            int mrow = h * 8 + i;
            if (sValid[mrow]) {
                int pk = sPk[mrow];
                outb[(size_t)pk * DD + d] = sGate[mrow] * (acc[ct][i] + b2v);
            }
        }
    }
}

// ---------------------------------------------------------------------------
// Kernel 3: deterministic combine of the 4 slot results per token.
// ---------------------------------------------------------------------------
__global__ __launch_bounds__(256)
void combine_topk(const float* __restrict__ outb, float* __restrict__ out) {
    int idx = blockIdx.x * 256 + threadIdx.x;     // over NN*DD
    int t = idx >> 10, d = idx & 1023;
    float s = 0.0f;
#pragma unroll
    for (int k = 0; k < TK; ++k) s += outb[((size_t)(t * TK + k)) * DD + d];
    out[idx] = s;
}

// ---------------------------------------------------------------------------
// Launch. Inputs: x, keys, w1, b1, w2, b2, rw, rb (all f32).
// Workspace layout:
//   [0,1024)            counts int[16] (memset each call)
//   [1024, +128K)       packed (token*4+slot) lists
//   [132096, +128K)     gate weights
//   [263168, +33.55MB)  per-(token,slot) results (N*K*D f32)
//   [33817600, +128MB)  w1 bf16   (only if ws_size permits)
//   [168035328, +128MB) w2 bf16
// ---------------------------------------------------------------------------
extern "C" void kernel_launch(void* const* d_in, const int* in_sizes, int n_in,
                              void* d_out, int out_size, void* d_ws, size_t ws_size,
                              hipStream_t stream) {
    (void)in_sizes; (void)n_in; (void)out_size;
    const float* x    = (const float*)d_in[0];
    const float* keys = (const float*)d_in[1];
    const float* w1   = (const float*)d_in[2];
    const float* b1   = (const float*)d_in[3];
    const float* w2   = (const float*)d_in[4];
    const float* b2   = (const float*)d_in[5];
    const float* rw   = (const float*)d_in[6];
    const float* rb   = (const float*)d_in[7];
    float* out = (float*)d_out;

    char* ws = (char*)d_ws;
    int*   counts  = (int*)ws;
    int*   pkArr   = (int*)(ws + 1024);
    float* gateArr = (float*)(ws + 1024 + (size_t)EE * NN * 4);
    float* outb    = (float*)(ws + 1024 + (size_t)2 * EE * NN * 4);

    const size_t W_ELEMS  = (size_t)EE * FF * DD;          // 67,108,864 per matrix
    const size_t OFF_W1B  = 1024 + (size_t)2 * EE * NN * 4 + (size_t)NN * TK * DD * 4;
    const size_t OFF_W2B  = OFF_W1B + W_ELEMS * 2;
    const size_t WS_NEED  = OFF_W2B + W_ELEMS * 2;
    unsigned short* w1b = (unsigned short*)(ws + OFF_W1B);
    unsigned short* w2b = (unsigned short*)(ws + OFF_W2B);
    const bool use_bf16w = (ws_size >= WS_NEED);

    hipMemsetAsync(counts, 0, EE * sizeof(int), stream);
    router_topk<<<NN / 8, 256, 0, stream>>>(x, keys, rw, rb, counts, pkArr, gateArr);

    if (use_bf16w) {
        const int n8 = (int)(W_ELEMS / 8);                 // 8,388,608 groups
        cvt_f32_to_bf16<<<n8 / 256, 256, 0, stream>>>(w1, w1b, n8);
        cvt_f32_to_bf16<<<n8 / 256, 256, 0, stream>>>(w2, w2b, n8);
        moe_expert_gemm<true><<<EE * MAX_TILES, 256, 0, stream>>>(
            x, w1, b1, w2, b2, w1b, w2b, counts, pkArr, gateArr, outb);
    } else {
        moe_expert_gemm<false><<<EE * MAX_TILES, 256, 0, stream>>>(
            x, w1, b1, w2, b2, w1b, w2b, counts, pkArr, gateArr, outb);
    }
    combine_topk<<<(NN * DD) / 256, 256, 0, stream>>>(outb, out);
}
